// RobotDynamicsLSTM_11665131176376
// MI455X (gfx1250) — matrix operand order repaired and checked by reading, not compiled, not run on hardware
//
#include <hip/hip_runtime.h>
#include <hip/hip_bf16.h>

typedef __attribute__((ext_vector_type(16))) _Float16 v16h;
typedef __attribute__((ext_vector_type(8)))  _Float16 v8h;
typedef __attribute__((ext_vector_type(8)))  float    v8f;
typedef __attribute__((ext_vector_type(4)))  unsigned int v4u;
typedef __attribute__((ext_vector_type(8)))  int      v8i;
typedef __attribute__((ext_vector_type(4)))  int      v4i;

namespace {
constexpr int Bv = 2048, Tv = 1024, Iv = 5, Hv = 64, Ov = 3;
constexpr int NKC  = 3;        // augmented K = 96 = 3 chunks of 32
constexpr int AROW = 104;      // halfs per A row (96 + pad, 208B, 16B-aligned, conflict-skewed)
constexpr int GSTR = 260;      // floats per gate row (256 + pad, 16B-aligned, conflict-skewed)

#if __has_builtin(__builtin_amdgcn_rcpf)
__device__ __forceinline__ float fast_rcp(float v) { return __builtin_amdgcn_rcpf(v); }
#else
__device__ __forceinline__ float fast_rcp(float v) { return 1.0f / v; }
#endif

// sigmoid / tanh via v_exp_f32 + v_rcp_f32 (no IEEE division sequence)
__device__ __forceinline__ float sigm(float v)      { return fast_rcp(1.0f + __expf(-v)); }
__device__ __forceinline__ float tanh_fast(float v) { return 2.0f * fast_rcp(1.0f + __expf(-2.0f * v)) - 1.0f; }

#if __has_builtin(__builtin_amdgcn_tensor_load_to_lds)
#define USE_TDM 1
#else
#define USE_TDM 0
#endif

#if USE_TDM
// Issue one TDM 2D tile load: 16 rows x 5 f32, row stride Tv*Iv elements,
// packed contiguously into LDS at lds_off. Tracked by TENSORcnt.
__device__ __forceinline__ void tdm_fetch_x(const float* gptr, unsigned lds_off) {
  const unsigned long long ga = (unsigned long long)(uintptr_t)gptr;
  v4u g0;
  g0[0] = 1u;                                            // count=1 (valid), user mode
  g0[1] = lds_off;                                       // LDS byte address
  g0[2] = (unsigned)(ga & 0xFFFFFFFFu);                  // global_addr[31:0]
  g0[3] = (unsigned)((ga >> 32) & 0x1FFFFFFu) | (2u << 30); // global_addr[56:32] | type=2
  v8i g1;
  g1[0] = (2 << 16);        // workgroup_mask=0, data_size=2 (4 bytes)
  g1[1] = (Iv << 16);       // tensor_dim0 low16 = 5 (bits 63:48)
  g1[2] = (16 << 16);       // tensor_dim0 hi = 0, tensor_dim1 low16 = 16
  g1[3] = (Iv << 16);       // tensor_dim1 hi = 0, tile_dim0 = 5
  g1[4] = 16;               // tile_dim1 = 16, tile_dim2 = 0
  g1[5] = Tv * Iv;          // tensor_dim0_stride[31:0] = 5120 elements
  g1[6] = 0;                // stride hi16 = 0, tensor_dim1_stride lo = 0
  g1[7] = 0;
  v4i z4 = {0, 0, 0, 0};
#if __clang_major__ >= 23
  v8i z8 = {0, 0, 0, 0, 0, 0, 0, 0};
  __builtin_amdgcn_tensor_load_to_lds(g0, g1, z4, z4, z8, 0);
#else
  __builtin_amdgcn_tensor_load_to_lds(g0, g1, z4, z4, 0);
#endif
}
#endif
} // namespace

// One workgroup = 16 batch rows through all T timesteps.
// Per step: gates[16x256] = A[16x96] x Baug[96x256] via v_wmma_f32_16x16x32_f16,
// where A = [h | x_t | 1 | 0] and Baug = [W_hh^T ; W_ih^T ; b_ih+b_hh ; 0].
// x_t tiles are prefetched by the Tensor Data Mover (double-buffered in LDS).
__global__ __launch_bounds__(256) void lstm_wmma(
    const float* __restrict__ x,    // [B,T,I]
    const float* __restrict__ Wih,  // [4H,I]
    const float* __restrict__ Whh,  // [4H,H]
    const float* __restrict__ bih,  // [4H]
    const float* __restrict__ bhh,  // [4H]
    const float* __restrict__ Wfc,  // [O,H]
    const float* __restrict__ bfc,  // [O]
    float* __restrict__ out)        // [B,O]
{
  __shared__ __align__(16) _Float16 As[16 * AROW];   // augmented A (f16)
  __shared__ __align__(16) float    gs[16 * GSTR];   // gate pre-activations (f32)
  __shared__ __align__(16) float    hf[16 * Hv];     // final h (f32) for the FC head
  __shared__ __align__(16) float    xstage[2][16 * Iv]; // TDM double buffer (80 f32 each)

  const int tid  = threadIdx.x;
  const int lane = tid & 31;
  const int wave = tid >> 5;          // 8 waves
  const int r0   = blockIdx.x * 16;   // batch-row base

  const int khi   = lane >> 4;        // which half of the wave
  const int ncol  = lane & 15;        // N (column-within-tile) for B and C/D
  const int mbase = khi * 8;          // C/D: vgpr r holds M = mbase + r

  // ---- Gather per-lane B-register images once (contiguous k order, same as the A loads):
  //      lane<16: b[j] = Baug[kc*32 + j][g]; lane>=16: b[j] = Baug[kc*32+16+j][g]
  v16h breg[2][NKC];
#pragma unroll
  for (int s = 0; s < 2; ++s) {
    const int g = (wave * 2 + s) * 16 + ncol;
#pragma unroll
    for (int kc = 0; kc < NKC; ++kc) {
      v16h bb;
#pragma unroll
      for (int j = 0; j < 16; ++j) {
        const int k = kc * 32 + (khi << 4) + j;   // augmented K index 0..95
        float v;
        if      (k < Hv)       v = Whh[g * Hv + k];
        else if (k < Hv + Iv)  v = Wih[g * Iv + (k - Hv)];
        else if (k == Hv + Iv) v = bih[g] + bhh[g];
        else                   v = 0.0f;
        bb[j] = (_Float16)v;
      }
      breg[s][kc] = bb;
    }
  }

  const float* xbase = x + (size_t)r0 * Tv * Iv;   // tile start row: x[r0][t][0]

  // ---- Init A image: h = 0, pad = 0; constant-1 column at k = H+I (bias row).
  for (int e = tid; e < 16 * AROW; e += 256) As[e] = (_Float16)0.0f;
  __syncthreads();
  if (tid < 16) As[tid * AROW + Hv + Iv] = (_Float16)1.0f;

#if USE_TDM
  if (wave == 0) {
    tdm_fetch_x(xbase, (unsigned)(uintptr_t)&xstage[0][0]);   // prefetch t=0
    __builtin_amdgcn_s_wait_tensorcnt(0);
  }
#else
  if (tid < 16 * Iv) xstage[0][tid] = xbase[(size_t)(tid / Iv) * Tv * Iv + (tid % Iv)];
#endif
  __syncthreads();

  // ---- c state: thread owns 4 consecutive hidden elems of row um.
  const int um = tid >> 4;
  const int uk = (tid & 15) * 4;
  float4 cst = make_float4(0.0f, 0.0f, 0.0f, 0.0f);

  const int arow = ncol * AROW;       // A-matrix: row M = lane & 15
  const int xm = tid / Iv, xi = tid % Iv;

  for (int t = 0; t < Tv; ++t) {
    // stage x_t (from TDM buffer) into A image; kick off prefetch of x_{t+1}
    if (tid < 16 * Iv) As[xm * AROW + Hv + xi] = (_Float16)xstage[t & 1][tid];
#if USE_TDM
    if (wave == 0 && t + 1 < Tv)
      tdm_fetch_x(xbase + (size_t)(t + 1) * Iv,
                  (unsigned)(uintptr_t)&xstage[(t + 1) & 1][0]);
#else
    if (t + 1 < Tv && tid < 16 * Iv)
      xstage[(t + 1) & 1][tid] =
          xbase[(size_t)(tid / Iv) * Tv * Iv + (size_t)(t + 1) * Iv + (tid % Iv)];
#endif
    __syncthreads();

    // ---- Load A chunks (contiguous k = 16*khi + j order: two contiguous 16B runs/lane)
    v16h a[NKC];
#pragma unroll
    for (int kc = 0; kc < NKC; ++kc) {
      const v8h lo = *(const v8h*)(&As[arow + kc * 32 + 16 * khi]);
      const v8h hi = *(const v8h*)(&As[arow + kc * 32 + 16 * khi + 8]);
      a[kc] = __builtin_shufflevector(lo, hi, 0, 1, 2, 3, 4, 5, 6, 7,
                                              8, 9, 10, 11, 12, 13, 14, 15);
    }

    // ---- 2 N-tiles per wave, 3 K-chunks each: 6 WMMAs / wave / step
#pragma unroll
    for (int s = 0; s < 2; ++s) {
      v8f acc = {};
      acc = __builtin_amdgcn_wmma_f32_16x16x32_f16(false, a[0], false, breg[s][0],
                                                   (short)0, acc, false, false);
      acc = __builtin_amdgcn_wmma_f32_16x16x32_f16(false, a[1], false, breg[s][1],
                                                   (short)0, acc, false, false);
      acc = __builtin_amdgcn_wmma_f32_16x16x32_f16(false, a[2], false, breg[s][2],
                                                   (short)0, acc, false, false);
      const int g = (wave * 2 + s) * 16 + ncol;
#pragma unroll
      for (int r = 0; r < 8; ++r)
        gs[(mbase + r) * GSTR + g] = acc[r];
    }
    __syncthreads();

    // ---- Gate update: i,f,g,o at columns k, H+k, 2H+k, 3H+k
    {
      const float* gr = &gs[um * GSTR + uk];
      const float4 ig = *(const float4*)(gr);
      const float4 fg = *(const float4*)(gr + Hv);
      const float4 gg = *(const float4*)(gr + 2 * Hv);
      const float4 og = *(const float4*)(gr + 3 * Hv);

      float4 hn;
      cst.x = sigm(fg.x) * cst.x + sigm(ig.x) * tanh_fast(gg.x);
      cst.y = sigm(fg.y) * cst.y + sigm(ig.y) * tanh_fast(gg.y);
      cst.z = sigm(fg.z) * cst.z + sigm(ig.z) * tanh_fast(gg.z);
      cst.w = sigm(fg.w) * cst.w + sigm(ig.w) * tanh_fast(gg.w);
      hn.x = sigm(og.x) * tanh_fast(cst.x);
      hn.y = sigm(og.y) * tanh_fast(cst.y);
      hn.z = sigm(og.z) * tanh_fast(cst.z);
      hn.w = sigm(og.w) * tanh_fast(cst.w);

      _Float16* hw = &As[um * AROW + uk];
      hw[0] = (_Float16)hn.x; hw[1] = (_Float16)hn.y;
      hw[2] = (_Float16)hn.z; hw[3] = (_Float16)hn.w;

      if (t == Tv - 1) {
        float* hfp = &hf[um * Hv + uk];
        hfp[0] = hn.x; hfp[1] = hn.y; hfp[2] = hn.z; hfp[3] = hn.w;
      }
    }

#if USE_TDM
    if (wave == 0) __builtin_amdgcn_s_wait_tensorcnt(0);  // x_{t+1} landed
#endif
    __syncthreads();
  }

  // ---- FC head: pred[m, o] = h_last[m,:] . W_fc[o,:] + b_fc[o]
  if (tid < 16 * Ov) {
    const int m = tid / Ov, o = tid % Ov;
    float s = bfc[o];
#pragma unroll 8
    for (int k = 0; k < Hv; ++k) s += hf[m * Hv + k] * Wfc[o * Hv + k];
    out[(size_t)(r0 + m) * Ov + o] = s;
  }
}

extern "C" void kernel_launch(void* const* d_in, const int* in_sizes, int n_in,
                              void* d_out, int out_size, void* d_ws, size_t ws_size,
                              hipStream_t stream) {
  (void)in_sizes; (void)n_in; (void)d_ws; (void)ws_size; (void)out_size;
  const float* x   = (const float*)d_in[0];
  const float* Wih = (const float*)d_in[1];
  const float* Whh = (const float*)d_in[2];
  const float* bih = (const float*)d_in[3];
  const float* bhh = (const float*)d_in[4];
  const float* Wfc = (const float*)d_in[5];
  const float* bfc = (const float*)d_in[6];
  float* out = (float*)d_out;

  dim3 grid(Bv / 16), block(256);
  hipLaunchKernelGGL(lstm_wmma, grid, block, 0, stream,
                     x, Wih, Whh, bih, bhh, Wfc, bfc, out);
}
